// GraphAttentionLayer_85968065397415
// MI455X (gfx1250) — matrix operand order repaired and checked by reading, not compiled, not run on hardware
//
#include <hip/hip_runtime.h>
#include <stdint.h>

#define BB 8
#define NN 2048
#define FF 256

typedef __attribute__((ext_vector_type(16))) __bf16 v16bf;
typedef __attribute__((ext_vector_type(8)))  float v8f;
typedef __attribute__((ext_vector_type(8)))  unsigned int v8u;

// workspace byte offsets
#define WP_OFF   0u        // packed W (B-fragment layout), 256*256 bf16 = 128 KB
#define WA_OFF   131072u   // wa1[256], wa2[256] f32 = 2 KB
#define S1_OFF   133120u   // s1[8*2048] f32 = 64 KB (pre-scaled by log2 e)
#define S2_OFF   198656u   // s2[8*2048] f32 = 64 KB (pre-scaled by log2 e)
#define ADJ_OFF  264192u   // adj bitmask, 2048*64 u32 = 512 KB
#define HB_OFF   788480u   // h in B-fragment bf16 layout, 16384*256*2 = 8 MB

#define LOG2E 1.44269504088896340736f

// pack two f32 into one dword of bf16 (round-half-up): 2 adds + 1 v_perm_b32
__device__ __forceinline__ unsigned int pkbf(float lo, float hi){
  unsigned int ul = __builtin_bit_cast(unsigned int, lo) + 0x8000u;
  unsigned int uh = __builtin_bit_cast(unsigned int, hi) + 0x8000u;
  return __builtin_amdgcn_perm(uh, ul, 0x07060302u); // [uh.hi16 : ul.hi16]
}
__device__ __forceinline__ unsigned short f2bf(float f){
  unsigned int u = __builtin_bit_cast(unsigned int, f) + 0x8000u;
  return (unsigned short)(u >> 16);
}
__device__ __forceinline__ v8f zero8(){ v8f z = {0.f,0.f,0.f,0.f,0.f,0.f,0.f,0.f}; return z; }

// ---- pack W into WMMA B-fragment layout: block(kt,nt): lane = col%16 + 16*((k%16)>=8), slot = k%8 + 8*(k%32>=16)
__global__ void k_prep_w(const float* __restrict__ W, unsigned short* __restrict__ Wp){
  int k = blockIdx.x, c = threadIdx.x;
  float v = W[k*FF + c];
  int kt = k >> 5, kin = k & 31, nt = c >> 4;
  int lane = (c & 15) | (((kin >> 3) & 1) << 4);
  int slot = (kin & 7) | ((kin >> 4) << 3);
  Wp[((kt*16 + nt)*32 + lane)*16 + slot] = f2bf(v);
}

// ---- adj -> bitmask: adjB[row][word] bit j = adj[row][word*32+j] != 0
__global__ void k_prep_adj(const int* __restrict__ adj, unsigned int* __restrict__ adjB){
  int row = blockIdx.x, wrd = threadIdx.x;
  const int* ar = adj + (size_t)row*NN + wrd*32;
  unsigned int bits = 0u;
  #pragma unroll
  for (int j = 0; j < 32; j++) bits |= (ar[j] != 0 ? 1u : 0u) << j;
  adjB[row*64 + wrd] = bits;
}

// ---- wa1 = W @ a1, wa2 = W @ a2 (256-vectors)
__global__ void k_wa(const float* __restrict__ W, const float* __restrict__ a,
                     float* __restrict__ wa){
  int f = threadIdx.x;
  float t1 = 0.f, t2 = 0.f;
  for (int o = 0; o < FF; o++){
    float wv = W[f*FF + o];
    t1 += wv * a[o];
    t2 += wv * a[FF + o];
  }
  wa[f] = t1; wa[FF + f] = t2;
}

// ---- s1[b,n] = log2e * x[b,n,:]·wa1 ; s2 likewise (one wave per row, fp32)
__global__ void k_s(const float* __restrict__ x, const float* __restrict__ wa,
                    float* __restrict__ s1, float* __restrict__ s2){
  int lane = threadIdx.x & 31;
  int row  = blockIdx.x*4 + (threadIdx.x >> 5);
  const float* xr = x + (size_t)row*FF;
  float p1 = 0.f, p2 = 0.f;
  #pragma unroll
  for (int i = 0; i < 8; i++){
    int c = lane + 32*i;
    float xv = xr[c];
    p1 += xv*wa[c]; p2 += xv*wa[FF + c];
  }
  #pragma unroll
  for (int off = 16; off >= 1; off >>= 1){
    p1 += __shfl_xor(p1, off, 32);
    p2 += __shfl_xor(p2, off, 32);
  }
  if (lane == 0){ s1[row] = p1*LOG2E; s2[row] = p2*LOG2E; }
}

// ---- h = x @ W via WMMA bf16; store h directly in B-fragment layout for the attention GEMM
__global__ void __launch_bounds__(128) k_h(const float* __restrict__ x,
                                           const unsigned char* __restrict__ Wp,
                                           unsigned char* __restrict__ hB){
  int tid = threadIdx.x, lane = tid & 31, w = tid >> 5;
  int i0 = blockIdx.x * 32;          // 32 rows per block (one hB k-tile)
  int ntbase = w*4;                  // each wave: 64 output features
  v8f acc[2][4];
  for (int t = 0; t < 2; t++) for (int ct = 0; ct < 4; ct++) acc[t][ct] = zero8();

  for (int kb = 0; kb < 8; kb++){
    v16bf af[2];
    #pragma unroll
    for (int t = 0; t < 2; t++){
      int row   = i0 + t*16 + (lane & 15);
      int kbase = kb*32 + ((lane < 16) ? 0 : 8);
      const float4* x4 = reinterpret_cast<const float4*>(x + (size_t)row*FF + kbase);
      float4 r0 = x4[0], r1 = x4[1], r2 = x4[4], r3 = x4[5]; // K {0..7} and {16..23} rel. to kbase
      v8u u;
      u[0]=pkbf(r0.x,r0.y); u[1]=pkbf(r0.z,r0.w);
      u[2]=pkbf(r1.x,r1.y); u[3]=pkbf(r1.z,r1.w);
      u[4]=pkbf(r2.x,r2.y); u[5]=pkbf(r2.z,r2.w);
      u[6]=pkbf(r3.x,r3.y); u[7]=pkbf(r3.z,r3.w);
      af[t] = __builtin_bit_cast(v16bf, u);
    }
    #pragma unroll
    for (int ct = 0; ct < 4; ct++){
      int nt = ntbase + ct;
      const uint4* bp = reinterpret_cast<const uint4*>(Wp + (size_t)((kb*16 + nt)*32 + lane)*32);
      uint4 b0 = bp[0], b1 = bp[1];
      v8u bu; bu[0]=b0.x;bu[1]=b0.y;bu[2]=b0.z;bu[3]=b0.w;bu[4]=b1.x;bu[5]=b1.y;bu[6]=b1.z;bu[7]=b1.w;
      v16bf bfrag = __builtin_bit_cast(v16bf, bu);
      #pragma unroll
      for (int t = 0; t < 2; t++)
        acc[t][ct] = __builtin_amdgcn_wmma_f32_16x16x32_bf16(
            false, af[t], false, bfrag, (short)0, acc[t][ct], false, false);
    }
  }
  // D tile: lane holds 8 consecutive rows of one column -> pack & store straight into B-layout
  #pragma unroll
  for (int t = 0; t < 2; t++){
    #pragma unroll
    for (int ct = 0; ct < 4; ct++){
      int nt = ntbase + ct;
      uint4 d;
      d.x = pkbf(acc[t][ct][0], acc[t][ct][1]);
      d.y = pkbf(acc[t][ct][2], acc[t][ct][3]);
      d.z = pkbf(acc[t][ct][4], acc[t][ct][5]);
      d.w = pkbf(acc[t][ct][6], acc[t][ct][7]);
      size_t addr = (size_t)blockIdx.x*16384 + (size_t)nt*1024
                  + (size_t)((lane & 15) + 16*(lane >> 4))*32 + (size_t)t*16;
      *reinterpret_cast<uint4*>(hB + addr) = d;
    }
  }
}

// ---- fused masked attention + out = softmax(e) @ h, flash style, one wave = 16 rows x 256 feats
__global__ void __launch_bounds__(128) k_attn(const unsigned int* __restrict__ adjB,
                                              const float* __restrict__ s1g,
                                              const float* __restrict__ s2g,
                                              const unsigned char* __restrict__ hB,
                                              float* __restrict__ out){
  int tid = threadIdx.x, lane = tid & 31, w = tid >> 5;
  int b   = blockIdx.x >> 5;                 // 32 blocks per batch
  int grp = blockIdx.x & 31;
  int i0  = grp*64 + w*16;                   // row base within batch
  int r   = i0 + (lane & 15);
  float s1v = s1g[b*NN + r];
  const float* s2r = s2g + b*NN;
  const unsigned int* abr = adjB + (size_t)r*64;
  int hh  = (lane >> 4) * 8;
  int off = (lane < 16) ? 0 : 8;             // A-fragment K slot base for this lane

  float mrow = -1e30f, lrow = 0.f;
  v8f acc[16];
  #pragma unroll
  for (int nt = 0; nt < 16; nt++) acc[nt] = zero8();

  for (int ks = 0; ks < 64; ks++){
    int kb = ks*32 + off;
    const float4* s2p = reinterpret_cast<const float4*>(s2r + kb);
    float4 q0 = s2p[0], q1 = s2p[1], q2 = s2p[4], q3 = s2p[5];
    unsigned int bs = abr[ks] >> off;        // bits j (j<8) and j+8 (j>=8) are our 16 K slots
    float sv[16] = {q0.x,q0.y,q0.z,q0.w, q1.x,q1.y,q1.z,q1.w,
                    q2.x,q2.y,q2.z,q2.w, q3.x,q3.y,q3.z,q3.w};
    float e[16];
    #pragma unroll
    for (int j = 0; j < 16; j++){
      unsigned int msk = (j < 8) ? (1u << j) : (1u << (j + 8));
      float v = s1v + sv[j];
      v = fmaxf(v, 0.2f*v);                  // leaky_relu(0.2), scale-invariant
      e[j] = (bs & msk) ? v : (v - 1.4426950e9f);   // -1e9 * log2(e)
    }
    float lm = e[0];
    #pragma unroll
    for (int j = 1; j < 16; j++) lm = fmaxf(lm, e[j]);
    float om   = fmaxf(lm, __shfl_xor(lm, 16, 32));
    float mold = mrow;
    float mnew = fmaxf(mold, om);
    mrow = mnew;
    float ps = 0.f, p[16];
    #pragma unroll
    for (int j = 0; j < 16; j++){ p[j] = __builtin_amdgcn_exp2f(e[j] - mnew); ps += p[j]; }
    ps += __shfl_xor(ps, 16, 32);

    if (__any(mnew > mold)){                 // row max moved: rescale state + accumulators
      float scale = __builtin_amdgcn_exp2f(mold - mnew);
      lrow = lrow*scale + ps;
      float sc[8];
      #pragma unroll
      for (int g = 0; g < 8; g++) sc[g] = __shfl(scale, hh + g, 32);
      #pragma unroll
      for (int nt = 0; nt < 16; nt++){
        #pragma unroll
        for (int g = 0; g < 8; g++) acc[nt][g] *= sc[g];
      }
    } else {
      lrow += ps;
    }

    v8u pu;
    #pragma unroll
    for (int j = 0; j < 8; j++) pu[j] = pkbf(p[2*j], p[2*j+1]);
    v16bf af = __builtin_bit_cast(v16bf, pu);

    size_t hbase = ((size_t)(b*64 + ks))*16384 + (size_t)lane*32;
    #pragma unroll
    for (int nt = 0; nt < 16; nt++){
      const uint4* hp = reinterpret_cast<const uint4*>(hB + hbase + (size_t)nt*1024);
      uint4 h0 = hp[0], h1 = hp[1];
      v8u hu; hu[0]=h0.x;hu[1]=h0.y;hu[2]=h0.z;hu[3]=h0.w;hu[4]=h1.x;hu[5]=h1.y;hu[6]=h1.z;hu[7]=h1.w;
      v16bf hf = __builtin_bit_cast(v16bf, hu);
      acc[nt] = __builtin_amdgcn_wmma_f32_16x16x32_bf16(
          false, af, false, hf, (short)0, acc[nt], false, false);
    }
  }

  float linv[8];
  #pragma unroll
  for (int g = 0; g < 8; g++){ float lv = __shfl(lrow, hh + g, 32); linv[g] = 1.0f/lv; }
  #pragma unroll
  for (int nt = 0; nt < 16; nt++){
    #pragma unroll
    for (int g = 0; g < 8; g++){
      int row = i0 + 8*(lane >> 4) + g;
      int col = nt*16 + (lane & 15);
      out[((size_t)b*NN + row)*FF + col] = acc[nt][g] * linv[g];
    }
  }
}

extern "C" void kernel_launch(void* const* d_in, const int* in_sizes, int n_in,
                              void* d_out, int out_size, void* d_ws, size_t ws_size,
                              hipStream_t stream){
  (void)in_sizes; (void)n_in; (void)out_size; (void)ws_size;
  const float* x   = (const float*)d_in[0];
  const float* W   = (const float*)d_in[1];
  const float* a   = (const float*)d_in[2];
  const int*   adj = (const int*)d_in[3];
  float* out = (float*)d_out;
  unsigned char* ws = (unsigned char*)d_ws;
  unsigned short* Wp  = (unsigned short*)(ws + WP_OFF);
  float*         wa  = (float*)(ws + WA_OFF);
  float*         s1  = (float*)(ws + S1_OFF);
  float*         s2  = (float*)(ws + S2_OFF);
  unsigned int*  adb = (unsigned int*)(ws + ADJ_OFF);
  unsigned char* hB  = ws + HB_OFF;

  k_prep_w  <<<256, 256, 0, stream>>>(W, Wp);
  k_prep_adj<<<NN, 64, 0, stream>>>(adj, adb);
  k_wa      <<<1,   256, 0, stream>>>(W, a, wa);
  k_s       <<<BB*NN/4, 128, 0, stream>>>(x, wa, s1, s2);
  k_h       <<<BB*NN/32, 128, 0, stream>>>(x, (const unsigned char*)(ws + WP_OFF), hB);
  k_attn    <<<BB*NN/64, 128, 0, stream>>>(adb, s1, s2, hB, out);
}
